// CausalSelfAttention_56444460204244
// MI455X (gfx1250) — matrix operand order repaired and checked by reading, not compiled, not run on hardware
//
#include <hip/hip_runtime.h>
#include <stdint.h>

// ---------------------------------------------------------------------------
// CDNA5 (gfx1250) causal self-attention, bf16 WMMA with f32 accumulation.
// B=2, T=2048, C=1024, H=16, d=64 (hardcoded from the reference).
// ---------------------------------------------------------------------------

typedef __attribute__((ext_vector_type(16))) __bf16 v16bf;
typedef __attribute__((ext_vector_type(8)))  float  v8f;

#define WMMA_BF16(a, b, c) \
  __builtin_amdgcn_wmma_f32_16x16x32_bf16(false, (a), false, (b), (short)0, (c), false, false)

static __device__ __forceinline__ unsigned short f2bf(float f) {
  union { float f; uint32_t u; } x; x.f = f;
  uint32_t u = x.u;
  u += 0x7FFFu + ((u >> 16) & 1u);          // round to nearest even
  return (unsigned short)(u >> 16);
}

union FragU { v16bf v; uint32_t u[8]; };

// A-matrix 16x32 bf16 fragment (ISA 7.12.2): lane&15 = row M,
// lane>=16 shifts K by +8 (first 4 VGPRs) / +8 (last 4). K pairs contiguous.
static __device__ __forceinline__ v16bf load_a_frag(const unsigned short* A, int lda) {
  FragU r;
  int lane = threadIdx.x & 31;
  int row  = lane & 15;
  int koff = (lane >> 4) << 3;              // 0 or 8
  const unsigned short* base = A + row * lda + koff;
#pragma unroll
  for (int v = 0; v < 8; ++v) {
    int k = (v < 4) ? (2 * v) : (8 + 2 * v); // v>=4 -> 16 + 2*(v-4)
    r.u[v] = *(const uint32_t*)(base + k);
  }
  return r.v;
}

// B-matrix 32x16 bf16 fragment: lane&15 = column N, lane>=16 shifts K by +8;
// VGPR v<4 holds K pair 2v, v>=4 holds 16+2(v-4). Source stored N-major: Bt[n*ldb + k].
static __device__ __forceinline__ v16bf load_b_frag(const unsigned short* Bt, int ldb) {
  FragU r;
  int lane  = threadIdx.x & 31;
  int n     = lane & 15;
  int kbase = (lane >> 4) << 3;             // 0 or 8
  const unsigned short* base = Bt + n * ldb + kbase;
#pragma unroll
  for (int v = 0; v < 8; ++v) r.u[v] = *(const uint32_t*)(base + ((v < 4) ? (2 * v) : (8 + 2 * v)));
  return r.v;
}

// ---------------------------------------------------------------------------
// Precision conversion / weight transpose kernels
// ---------------------------------------------------------------------------
__global__ __launch_bounds__(256) void k_cvt_bf16(const float* __restrict__ in,
                                                  unsigned short* __restrict__ out, int n) {
  int i = blockIdx.x * blockDim.x + threadIdx.x;
  if (i < n) out[i] = f2bf(in[i]);
}

// W[K][N] (row-major) -> Wt[N][K] bf16
__global__ __launch_bounds__(256) void k_transpose_bf16(const float* __restrict__ W,
                                                        unsigned short* __restrict__ Wt,
                                                        int K, int N) {
  int idx = blockIdx.x * blockDim.x + threadIdx.x;  // idx = nn*K + kk
  if (idx >= K * N) return;
  int kk = idx & (K - 1);                           // K is a power of two (1024)
  int nn = idx / K;
  Wt[idx] = f2bf(W[(size_t)kk * N + nn]);
}

// ---------------------------------------------------------------------------
// QKV projection: xb[4096,1024] @ WqkvT[3072,1024]^T -> Q,K [B,H,T,64], V^T [B,H,64,T]
// One wave = 32x32 output tile (2x2 WMMA tiles).
// ---------------------------------------------------------------------------
__global__ __launch_bounds__(256) void k_gemm_qkv(const unsigned short* __restrict__ xb,
                                                  const unsigned short* __restrict__ wt,
                                                  unsigned short* __restrict__ q,
                                                  unsigned short* __restrict__ k,
                                                  unsigned short* __restrict__ vt) {
  int lane  = threadIdx.x & 31;
  int wid   = (blockIdx.x * blockDim.x + threadIdx.x) >> 5;
  int tileN = wid % 96;                 // 3072/32
  int tileM = wid / 96;                 // 4096/32
  int row0 = tileM * 32, col0 = tileN * 32;

  v8f acc00 = {}, acc01 = {}, acc10 = {}, acc11 = {};
  for (int kk = 0; kk < 1024; kk += 32) {
    v16bf a0 = load_a_frag(xb + (size_t)row0 * 1024 + kk, 1024);
    v16bf a1 = load_a_frag(xb + (size_t)(row0 + 16) * 1024 + kk, 1024);
    v16bf b0 = load_b_frag(wt + (size_t)col0 * 1024 + kk, 1024);
    v16bf b1 = load_b_frag(wt + (size_t)(col0 + 16) * 1024 + kk, 1024);
    acc00 = WMMA_BF16(a0, b0, acc00);
    acc01 = WMMA_BF16(a0, b1, acc01);
    acc10 = WMMA_BF16(a1, b0, acc10);
    acc11 = WMMA_BF16(a1, b1, acc11);
  }

  int n  = lane & 15;
  int mo = (lane >> 4) << 3;
#pragma unroll
  for (int ti = 0; ti < 2; ++ti) {
#pragma unroll
    for (int tj = 0; tj < 2; ++tj) {
      const v8f& a = ti ? (tj ? acc11 : acc10) : (tj ? acc01 : acc00);
#pragma unroll
      for (int r = 0; r < 8; ++r) {
        int grow = row0 + ti * 16 + mo + r;     // token index 0..4095
        int gcol = col0 + tj * 16 + n;          // 0..3071
        unsigned short bv = f2bf(a[r]);
        int b  = grow >> 11, t = grow & 2047;
        int wh = gcol >> 10;                    // 0=Q 1=K 2=V
        int c  = gcol & 1023;
        int h  = c >> 6, dd = c & 63;
        size_t bh = (size_t)(b * 16 + h);
        if (wh == 0)      q [(bh * 2048 + t) * 64 + dd] = bv;
        else if (wh == 1) k [(bh * 2048 + t) * 64 + dd] = bv;
        else              vt[(bh * 64 + dd) * 2048 + t] = bv;   // V transposed
      }
    }
  }
}

// ---------------------------------------------------------------------------
// Flash attention: one wave per 16-query tile, 32-key inner tiles, online
// softmax in the C/D register layout, P->A layout via LDS round trip.
// ---------------------------------------------------------------------------
__global__ __launch_bounds__(256) void k_attn(const unsigned short* __restrict__ q,
                                              const unsigned short* __restrict__ kmat,
                                              const unsigned short* __restrict__ vt,
                                              unsigned short* __restrict__ ob) {
  __shared__ unsigned short lds_p[8][16 * 32];   // per-wave 16x32 bf16 P tile
  int lane  = threadIdx.x & 31;
  int wv    = threadIdx.x >> 5;
  int chunk = blockIdx.x & 15;          // 16 chunks of 128 queries
  int bh    = blockIdx.x >> 4;          // 0..31
  int b = bh >> 4, h = bh & 15;
  int qt0 = chunk * 128 + wv * 16;

  const unsigned short* Q = q    + (size_t)bh * 2048 * 64;
  const unsigned short* K = kmat + (size_t)bh * 2048 * 64;
  const unsigned short* V = vt   + (size_t)bh * 64 * 2048;   // [64, T]

  v16bf aq0 = load_a_frag(Q + (size_t)qt0 * 64,      64);    // d 0..31
  v16bf aq1 = load_a_frag(Q + (size_t)qt0 * 64 + 32, 64);    // d 32..63

  v8f o0 = {}, o1 = {}, o2 = {}, o3 = {};
  float mrow[8], lrow[8];
#pragma unroll
  for (int r = 0; r < 8; ++r) { mrow[r] = -3.0e38f; lrow[r] = 0.0f; }

  int n  = lane & 15;
  int mo = (lane >> 4) << 3;
  unsigned short* pb = &lds_p[wv][0];

  for (int kt0 = 0; kt0 < qt0 + 16; kt0 += 32) {
    // S = Q K^T (d=64 -> two chained WMMAs per 16-key half)
    v16bf b0a = load_b_frag(K + (size_t)kt0 * 64,      64);
    v16bf b0b = load_b_frag(K + (size_t)kt0 * 64 + 32, 64);
    v8f s0 = {};
    s0 = WMMA_BF16(aq0, b0a, s0);
    s0 = WMMA_BF16(aq1, b0b, s0);
    v16bf b1a = load_b_frag(K + (size_t)(kt0 + 16) * 64,      64);
    v16bf b1b = load_b_frag(K + (size_t)(kt0 + 16) * 64 + 32, 64);
    v8f s1 = {};
    s1 = WMMA_BF16(aq0, b1a, s1);
    s1 = WMMA_BF16(aq1, b1b, s1);

    bool needMask = (kt0 + 31 > qt0);
#pragma unroll
    for (int r = 0; r < 8; ++r) {
      float e0 = s0[r] * 0.125f;                  // 1/sqrt(64)
      float e1 = s1[r] * 0.125f;
      if (needMask) {
        int tq = qt0 + mo + r;
        if (kt0 + n      > tq) e0 = -3.0e38f;
        if (kt0 + 16 + n > tq) e1 = -3.0e38f;
      }
      // row max across the 16 lanes holding this row
      float a = fmaxf(e0, e1);
      a = fmaxf(a, __shfl_xor(a, 1));
      a = fmaxf(a, __shfl_xor(a, 2));
      a = fmaxf(a, __shfl_xor(a, 4));
      a = fmaxf(a, __shfl_xor(a, 8));
      float mn   = fmaxf(mrow[r], a);
      float corr = __expf(mrow[r] - mn);
      float p0   = __expf(e0 - mn);
      float p1   = __expf(e1 - mn);
      float ps   = p0 + p1;
      ps += __shfl_xor(ps, 1);
      ps += __shfl_xor(ps, 2);
      ps += __shfl_xor(ps, 4);
      ps += __shfl_xor(ps, 8);
      lrow[r] = lrow[r] * corr + ps;
      mrow[r] = mn;
      o0[r] *= corr; o1[r] *= corr; o2[r] *= corr; o3[r] *= corr;
      // C-layout -> row-major 16x32 P tile in LDS (bf16)
      int m = mo + r;
      pb[m * 32 + n]      = f2bf(p0);
      pb[m * 32 + 16 + n] = f2bf(p1);
    }
    // reload P in A-fragment layout (same-wave DS ops stay in order)
    v16bf ap = load_a_frag(pb, 32);
    // O += P V ; V^T[64,T] gives contiguous-K B fragments
    v16bf bv0 = load_b_frag(V + (size_t)0  * 2048 + kt0, 2048);
    o0 = WMMA_BF16(ap, bv0, o0);
    v16bf bv1 = load_b_frag(V + (size_t)16 * 2048 + kt0, 2048);
    o1 = WMMA_BF16(ap, bv1, o1);
    v16bf bv2 = load_b_frag(V + (size_t)32 * 2048 + kt0, 2048);
    o2 = WMMA_BF16(ap, bv2, o2);
    v16bf bv3 = load_b_frag(V + (size_t)48 * 2048 + kt0, 2048);
    o3 = WMMA_BF16(ap, bv3, o3);
  }

  // normalize and scatter to [B, T, C] bf16 for the output projection
#pragma unroll
  for (int r = 0; r < 8; ++r) {
    float inv = 1.0f / lrow[r];
    int m = mo + r;
    size_t row = (size_t)(b * 2048 + qt0 + m) * 1024 + h * 64 + n;
    ob[row + 0]  = f2bf(o0[r] * inv);
    ob[row + 16] = f2bf(o1[r] * inv);
    ob[row + 32] = f2bf(o2[r] * inv);
    ob[row + 48] = f2bf(o3[r] * inv);
  }
}

// ---------------------------------------------------------------------------
// Output projection: ob[4096,1024] @ WprojT[1024,1024]^T -> out fp32
// ---------------------------------------------------------------------------
__global__ __launch_bounds__(256) void k_gemm_proj(const unsigned short* __restrict__ ob,
                                                   const unsigned short* __restrict__ wt,
                                                   float* __restrict__ out) {
  int lane  = threadIdx.x & 31;
  int wid   = (blockIdx.x * blockDim.x + threadIdx.x) >> 5;
  int tileN = wid & 31;                 // 1024/32
  int tileM = wid >> 5;                 // 4096/32
  int row0 = tileM * 32, col0 = tileN * 32;

  v8f acc00 = {}, acc01 = {}, acc10 = {}, acc11 = {};
  for (int kk = 0; kk < 1024; kk += 32) {
    v16bf a0 = load_a_frag(ob + (size_t)row0 * 1024 + kk, 1024);
    v16bf a1 = load_a_frag(ob + (size_t)(row0 + 16) * 1024 + kk, 1024);
    v16bf b0 = load_b_frag(wt + (size_t)col0 * 1024 + kk, 1024);
    v16bf b1 = load_b_frag(wt + (size_t)(col0 + 16) * 1024 + kk, 1024);
    acc00 = WMMA_BF16(a0, b0, acc00);
    acc01 = WMMA_BF16(a0, b1, acc01);
    acc10 = WMMA_BF16(a1, b0, acc10);
    acc11 = WMMA_BF16(a1, b1, acc11);
  }

  int n  = lane & 15;
  int mo = (lane >> 4) << 3;
#pragma unroll
  for (int ti = 0; ti < 2; ++ti) {
#pragma unroll
    for (int tj = 0; tj < 2; ++tj) {
      const v8f& a = ti ? (tj ? acc11 : acc10) : (tj ? acc01 : acc00);
#pragma unroll
      for (int r = 0; r < 8; ++r) {
        int grow = row0 + ti * 16 + mo + r;
        int gcol = col0 + tj * 16 + n;
        out[(size_t)grow * 1024 + gcol] = a[r];
      }
    }
  }
}

// ---------------------------------------------------------------------------
// Host launcher
// ---------------------------------------------------------------------------
extern "C" void kernel_launch(void* const* d_in, const int* in_sizes, int n_in,
                              void* d_out, int out_size, void* d_ws, size_t ws_size,
                              hipStream_t stream) {
  const float* x     = (const float*)d_in[0];   // [2,2048,1024]
  const float* Wqkv  = (const float*)d_in[1];   // [1024,3072]
  const float* Wproj = (const float*)d_in[2];   // [1024,1024]
  float* out = (float*)d_out;                   // [2,2048,1024] fp32

  unsigned short* ws     = (unsigned short*)d_ws;
  unsigned short* xb     = ws;                   // 4194304  x bf16
  unsigned short* wqkvT  = xb    + 4194304;      // 3145728  Wqkv^T bf16 [3072,1024]
  unsigned short* wprojT = wqkvT + 3145728;      // 1048576  Wproj^T bf16 [1024,1024]
  unsigned short* qb     = wprojT + 1048576;     // 4194304  Q [B,H,T,64]
  unsigned short* kb     = qb    + 4194304;      // 4194304  K [B,H,T,64]
  unsigned short* vtb    = kb    + 4194304;      // 4194304  V^T [B,H,64,T]
  unsigned short* ob     = vtb   + 4194304;      // 4194304  attn out [B,T,C]

  k_cvt_bf16      <<<4194304 / 256, 256, 0, stream>>>(x, xb, 4194304);
  k_transpose_bf16<<<3145728 / 256, 256, 0, stream>>>(Wqkv,  wqkvT, 1024, 3072);
  k_transpose_bf16<<<1048576 / 256, 256, 0, stream>>>(Wproj, wprojT, 1024, 1024);
  k_gemm_qkv      <<<1536, 256, 0, stream>>>(xb, wqkvT, qb, kb, vtb);
  k_attn          <<<512,  256, 0, stream>>>(qb, kb, vtb, ob);
  k_gemm_proj     <<<512,  256, 0, stream>>>(ob, wprojT, out);
}
